// W4A16B16O16Linear_40243843564263
// MI455X (gfx1250) — matrix operand order repaired and checked by reading, not compiled, not run on hardware
//
#include <hip/hip_runtime.h>

// W4A16 group-quantized GEMM for MI455X (gfx1250, wave32, WMMA + TDM).
// y[M,N] = x[M,K] @ ((q[K,N] - 8) * scale[K/128, N]),  fp32 acc, fp16 out.
//
// - 128x128 block tile, BK=32, 8 waves (4M x 2N), each wave 2x4 WMMA tiles.
// - A tile DMA'd global->LDS by the Tensor Data Mover (wave 0 issues the
//   descriptor; TDM padding recreates the 80B conflict-breaking pitch).
// - B tile: coalesced int32 loads, fp16 dequant, transposed LDS store.
// - Double-buffered LDS, one barrier per k-step; next-tile loads + TDM
//   overlap the current tile's v_wmma_f32_16x16x32_f16 chain.

typedef _Float16 v16h __attribute__((ext_vector_type(16)));
typedef _Float16 v8h  __attribute__((ext_vector_type(8)));
typedef float    v8f  __attribute__((ext_vector_type(8)));
typedef int      v4i  __attribute__((ext_vector_type(4)));
typedef int      v8i  __attribute__((ext_vector_type(8)));
typedef unsigned v4u  __attribute__((ext_vector_type(4)));

#define M_DIM 8192
#define K_DIM 4096
#define N_DIM 4096
#define BM 128
#define BN 128
#define BK 32
#define PITCH 40   // halves per LDS row (80 B = 64 B data + 16 B pad)

#if __has_builtin(__builtin_amdgcn_tensor_load_to_lds)
#define HAVE_TDM 1
#else
#define HAVE_TDM 0
#endif

__global__ __launch_bounds__(256) void w4a16_wmma_tdm_gemm(
    const _Float16* __restrict__ X,   // [M, K] fp16
    const int*      __restrict__ Q,   // [K, N] int4 codes in int32
    const _Float16* __restrict__ S,   // [K/128, N] fp16
    _Float16*       __restrict__ Y)   // [M, N] fp16
{
    __shared__ _Float16 sA[2][BM * PITCH];   // A tiles, row-major [m][k], padded pitch
    __shared__ _Float16 sB[2][BN * PITCH];   // dequantized B tiles, TRANSPOSED [n][k]

    const int t     = threadIdx.x;
    const int lane  = t & 31;
    const int wave  = t >> 5;       // 0..7
    const int waveM = wave & 3;     // 4 waves along M
    const int waveN = wave >> 2;    // 2 waves along N
    const int l16   = lane & 15;
    const int hi    = lane >> 4;

    const int bN = blockIdx.x * BN;
    const int bM = blockIdx.y * BM;

    // B staging: thread owns k-rows {2*kp, 2*kp+1}, n-cols nb..nb+7
    const int kp = t >> 4;          // 0..15
    const int nb = (t & 15) << 3;   // 0,8,...,120

    v8f acc[2][4];
    #pragma unroll
    for (int i = 0; i < 2; ++i)
        #pragma unroll
        for (int j = 0; j < 4; ++j)
            acc[i][j] = v8f{};

    // ---------- helpers ----------
    auto issue_A = [&](int k0, int buf) {
#if HAVE_TDM
        if (wave == 0) {
            // D# group 0: count=1 | lds_addr | 57-bit global addr | type=2
            unsigned lds = (unsigned)(unsigned long long)(&sA[buf][0]);
            unsigned long long ga =
                (unsigned long long)(const void*)(X + (size_t)bM * K_DIM + k0);
            v4u g0 = { 1u, lds, (unsigned)ga,
                       (unsigned)((ga >> 32) & 0x01FFFFFFull) | (2u << 30) };
            // D# group 1:
            //  data_size=1 (2B) | pad_enable | pad_interval=3 (16 DW = 64 B)
            //  | pad_amount=3 (4 DW = 16 B)  -> effective LDS pitch 80 B
            //  tensor_dim0=K, tensor_dim1=M, tile=32x128, dim0_stride=K
            v8i g1 = { (int)0x06D10000,
                       (int)((unsigned)(K_DIM & 0xFFFF) << 16),  // dim0[15:0]
                       (int)((unsigned)(M_DIM & 0xFFFF) << 16),  // dim1[15:0], dim0[31:16]=0
                       (int)(32u << 16),                         // tile_dim0=32
                       (int)128,                                 // tile_dim1=128
                       (int)K_DIM,                               // dim0_stride lo32
                       0, 0 };
            v4i gz4 = { 0, 0, 0, 0 };
            v8i gz8 = { 0, 0, 0, 0, 0, 0, 0, 0 };
            // clang-23 / therock lane: 6-arg form
            // (v4u g0, v8i g1, v4i g2, v4i g3, v8i extra, i32 cpol)
            __builtin_amdgcn_tensor_load_to_lds(g0, g1, gz4, gz4, gz8, 0);
        }
#else
        // Fallback: vector staging through registers
        #pragma unroll
        for (int i = 0; i < 2; ++i) {
            int c   = t + 256 * i;
            int row = c >> 2;
            int col = (c & 3) << 3;
            v8h a = *(const v8h*)(X + (size_t)(bM + row) * K_DIM + k0 + col);
            *(v8h*)(&sA[buf][row * PITCH + col]) = a;
        }
#endif
    };

    auto wait_A = [&]() {
#if HAVE_TDM
        if (wave == 0) {
#if __has_builtin(__builtin_amdgcn_s_wait_tensorcnt)
            __builtin_amdgcn_s_wait_tensorcnt(0);
#else
            asm volatile("s_wait_tensorcnt 0x0" ::: "memory");
#endif
        }
#endif
    };

    // ---------- prologue: stage tile 0 into buffer 0 ----------
    {
        issue_A(0, 0);
        const int* q0p = Q + (size_t)(2 * kp) * N_DIM + bN + nb;
        v4i q00 = *(const v4i*)(q0p);
        v4i q01 = *(const v4i*)(q0p + 4);
        v4i q10 = *(const v4i*)(q0p + N_DIM);
        v4i q11 = *(const v4i*)(q0p + N_DIM + 4);
        v8h sc  = *(const v8h*)(S + bN + nb);
        #pragma unroll
        for (int j = 0; j < 8; ++j) {
            int e0 = ((j < 4) ? q00[j] : q01[j - 4]) - 8;
            int e1 = ((j < 4) ? q10[j] : q11[j - 4]) - 8;
            _Float16 s = sc[j];
            union { _Float16 h[2]; unsigned u; } pk;
            pk.h[0] = (_Float16)e0 * s;
            pk.h[1] = (_Float16)e1 * s;
            *reinterpret_cast<unsigned*>(&sB[0][(nb + j) * PITCH + 2 * kp]) = pk.u;
        }
        wait_A();
        __syncthreads();
    }

    // ---------- main loop: one barrier per k-step, double-buffered ----------
    for (int k0 = 0; k0 < K_DIM; k0 += BK) {
        const int  cur  = (k0 >> 5) & 1;
        const int  nxt  = cur ^ 1;
        const bool more = (k0 + BK) < K_DIM;
        const int  kn   = k0 + BK;

        // Issue next tile's loads early so they run under the WMMA chain.
        v4i q00{}, q01{}, q10{}, q11{};
        v8h sc{};
        if (more) {
            issue_A(kn, nxt);
            const int* q0p = Q + (size_t)(kn + 2 * kp) * N_DIM + bN + nb;
            q00 = *(const v4i*)(q0p);
            q01 = *(const v4i*)(q0p + 4);
            q10 = *(const v4i*)(q0p + N_DIM);
            q11 = *(const v4i*)(q0p + N_DIM + 4);
            sc  = *(const v8h*)(S + (size_t)(kn >> 7) * N_DIM + bN + nb);
        }

        // ---- compute current tile: fragments + 2x4 WMMA ----
        // A 16x32 f16 frag: lanes 0-15 hold K0-7/K16-23; lanes 16-31 K8-15/K24-31
        v16h afr[2];
        #pragma unroll
        for (int i = 0; i < 2; ++i) {
            const _Float16* p = &sA[cur][(waveM * 32 + i * 16 + l16) * PITCH + hi * 8];
            v8h lo = *(const v8h*)p;
            v8h hh = *(const v8h*)(p + 16);
            afr[i] = __builtin_shufflevector(lo, hh,
                0,1,2,3,4,5,6,7,8,9,10,11,12,13,14,15);
        }
        // B 32x16 f16 frag: lanes 0-15 hold K0-7/K16-23 of column n; lanes 16-31 K8-15/K24-31
        v16h bfr[4];
        #pragma unroll
        for (int j = 0; j < 4; ++j) {
            const _Float16* p = &sB[cur][(waveN * 64 + j * 16 + l16) * PITCH + hi * 8];
            v8h lo = *(const v8h*)p;
            v8h hh = *(const v8h*)(p + 16);
            bfr[j] = __builtin_shufflevector(lo, hh,
                0,1,2,3,4,5,6,7,8,9,10,11,12,13,14,15);
        }
        #pragma unroll
        for (int i = 0; i < 2; ++i)
            #pragma unroll
            for (int j = 0; j < 4; ++j)
                acc[i][j] = __builtin_amdgcn_wmma_f32_16x16x32_f16(
                    false, afr[i], false, bfr[j],
                    (short)0, acc[i][j], false, false);

        // ---- dequant next B tile into the other buffer ----
        if (more) {
            #pragma unroll
            for (int j = 0; j < 8; ++j) {
                int e0 = ((j < 4) ? q00[j] : q01[j - 4]) - 8;
                int e1 = ((j < 4) ? q10[j] : q11[j - 4]) - 8;
                _Float16 s = sc[j];
                union { _Float16 h[2]; unsigned u; } pk;
                pk.h[0] = (_Float16)e0 * s;
                pk.h[1] = (_Float16)e1 * s;
                *reinterpret_cast<unsigned*>(&sB[nxt][(nb + j) * PITCH + 2 * kp]) = pk.u;
            }
            wait_A();
        }
        __syncthreads();
    }

    // ---------- epilogue: f32 acc -> fp16 ----------
    #pragma unroll
    for (int i = 0; i < 2; ++i) {
        #pragma unroll
        for (int j = 0; j < 4; ++j) {
            int rbase = bM + waveM * 32 + i * 16 + hi * 8;
            int cc    = bN + waveN * 64 + j * 16 + l16;
            #pragma unroll
            for (int r = 0; r < 8; ++r)
                Y[(size_t)(rbase + r) * N_DIM + cc] = (_Float16)acc[i][j][r];
        }
    }
}

extern "C" void kernel_launch(void* const* d_in, const int* in_sizes, int n_in,
                              void* d_out, int out_size, void* d_ws, size_t ws_size,
                              hipStream_t stream) {
    (void)in_sizes; (void)n_in; (void)out_size; (void)d_ws; (void)ws_size;
    const _Float16* X = (const _Float16*)d_in[0];
    const int*      Q = (const int*)d_in[1];
    const _Float16* S = (const _Float16*)d_in[2];
    _Float16*       Y = (_Float16*)d_out;

    dim3 grid(N_DIM / BN, M_DIM / BM);   // 32 x 64 = 2048 workgroups
    w4a16_wmma_tdm_gemm<<<grid, 256, 0, stream>>>(X, Q, S, Y);
}
